// TransformerBlock_61521111548414
// MI455X (gfx1250) — compile-verified, matrix operand order checked by reading, not run on hardware
//
#include <hip/hip_runtime.h>
#include <stdint.h>

// ---------------------------------------------------------------------------
// MI455X / gfx1250 transformer block.
//   B=2 S=2048 D=1024 F=4096 H=16 dk=64
// All matmuls: v_wmma_f32_16x16x32_bf16 (fp32 accumulate, bf16 operands).
// Global->LDS staging uses GLOBAL_LOAD_ASYNC_TO_LDS_B128 (ASYNCcnt path)
// when the toolchain declares the builtin, else load+ds_store fallback.
// ---------------------------------------------------------------------------

typedef __bf16 bf16_t;
typedef __attribute__((ext_vector_type(16))) __bf16 v16bf;
typedef __attribute__((ext_vector_type(8)))  float  v8f;
typedef __attribute__((ext_vector_type(4)))  int    v4i;

#if __has_builtin(__builtin_amdgcn_global_load_async_to_lds_b128)
#define HAVE_ASYNC_LDS 1
typedef __attribute__((address_space(1))) v4i* gas_v4i_p;
typedef __attribute__((address_space(3))) v4i* las_v4i_p;
#endif

static __device__ __forceinline__ bf16_t f2bf(float f) {
  union { float f; uint32_t u; } v; v.f = f;
  uint32_t r = v.u + 0x7FFFu + ((v.u >> 16) & 1u);   // round-to-nearest-even
  uint16_t h = (uint16_t)(r >> 16);
  bf16_t o; __builtin_memcpy(&o, &h, 2); return o;
}
static __device__ __forceinline__ float bf2f(bf16_t b) {
  uint16_t h; __builtin_memcpy(&h, &b, 2);
  union { uint32_t u; float f; } v; v.u = ((uint32_t)h) << 16; return v.f;
}
static __device__ __forceinline__ v8f zero8() {
  v8f z = {0.f,0.f,0.f,0.f,0.f,0.f,0.f,0.f}; return z;
}
static __device__ __forceinline__ v8f wmma_bf16(v16bf a, v16bf b, v8f c) {
  return __builtin_amdgcn_wmma_f32_16x16x32_bf16(false, a, false, b,
                                                 (short)0, c, false, false);
}

// A-matrix fragment (16 rows x 32 K), row-major in LDS, ld in halves.
// ISA 7.12.2: lane&15 = row; lane>=16 takes K+8; halves [0..7]=K..K+7,
// [8..15]=K+16..K+23.  Also used for B when the operand is stored [N,K].
static __device__ __forceinline__ v16bf frag_rowmajor(const bf16_t* p0, int ld, int lane) {
  const bf16_t* p = p0 + (lane & 15) * ld + ((lane >> 4) << 3);
  v16bf f;
#pragma unroll
  for (int i = 0; i < 8; ++i) { f[i] = p[i]; f[i + 8] = p[i + 16]; }
  return f;
}

// B-matrix fragment from K-major storage: tile[k][n] (e.g. V tile [key][dk]).
static __device__ __forceinline__ v16bf frag_kn(const bf16_t* base, int k0, int n0,
                                                int ld, int lane) {
  int n  = n0 + (lane & 15);
  int kb = k0 + ((lane >> 4) << 3);
  v16bf f;
#pragma unroll
  for (int i = 0; i < 8; ++i) {
    f[i]     = base[(kb + i)      * ld + n];
    f[i + 8] = base[(kb + 16 + i) * ld + n];
  }
  return f;
}

// Cooperative global->LDS tile copy, 16B (8 x bf16) per transaction.
static __device__ __forceinline__ void g2l_tile(bf16_t* lds, int ldl,
                                                const bf16_t* g, size_t ldg,
                                                int rows, int cols8,
                                                int tid, int nthr) {
  for (int idx = tid; idx < rows * cols8; idx += nthr) {
    int r = idx / cols8;
    int c = (idx - r * cols8) << 3;
#if defined(HAVE_ASYNC_LDS)
    __builtin_amdgcn_global_load_async_to_lds_b128(
        (gas_v4i_p)(g + (size_t)r * ldg + c),
        (las_v4i_p)(lds + r * ldl + c),
        0, 0);
#else
    *(uint4*)(lds + r * ldl + c) = *(const uint4*)(g + (size_t)r * ldg + c);
#endif
  }
}

// Completion fence for the async staging path (ASYNCcnt -> 0).
static __device__ __forceinline__ void wait_async() {
#if defined(HAVE_ASYNC_LDS)
#if __has_builtin(__builtin_amdgcn_s_wait_asynccnt)
  __builtin_amdgcn_s_wait_asynccnt(0);
#else
  asm volatile("s_wait_asynccnt 0" ::: "memory");
#endif
#endif
}

// ---------------------------------------------------------------------------
// fp32 -> bf16 conversion (weights), grid-stride
// ---------------------------------------------------------------------------
__global__ void k_cvt(const float* __restrict__ src, bf16_t* __restrict__ dst, size_t n) {
  size_t i = (size_t)blockIdx.x * blockDim.x + threadIdx.x;
  size_t stride = (size_t)gridDim.x * blockDim.x;
  for (; i < n; i += stride) dst[i] = f2bf(src[i]);
}

// ---------------------------------------------------------------------------
// RMSNorm: fp32 [rows, D] * w -> bf16.  One block (8 waves) per row.
// ---------------------------------------------------------------------------
__global__ __launch_bounds__(256) void k_rmsnorm(const float* __restrict__ X,
                                                 const float* __restrict__ w,
                                                 bf16_t* __restrict__ O, int D) {
  const int row = blockIdx.x;
  const float* x = X + (size_t)row * D;
  float ss = 0.f;
  for (int c = threadIdx.x; c < D; c += blockDim.x) { float v = x[c]; ss += v * v; }
#pragma unroll
  for (int m = 16; m >= 1; m >>= 1) ss += __shfl_xor(ss, m, 32);
  __shared__ float red[8];
  if ((threadIdx.x & 31) == 0) red[threadIdx.x >> 5] = ss;
  __syncthreads();
  float tot = 0.f;
#pragma unroll
  for (int i = 0; i < 8; ++i) tot += red[i];
  float r = rsqrtf(tot / (float)D + 1e-5f);
  for (int c = threadIdx.x; c < D; c += blockDim.x)
    O[(size_t)row * D + c] = f2bf(x[c] * r * w[c]);
}

// ---------------------------------------------------------------------------
// Generic GEMM: out[M,N] = A[M,K](bf16) x W[N,K](bf16)^T
//   outF != null : fp32 out (+ optional fp32 residual) ; else bf16 out
// 128x128 block tile, K-tile 64, 8 waves (2Mx4N), each wave 4x2 register tile
// of 16x16 WMMA -> 8 WMMAs per 6 fragment loads per 32-K step.
// M % 128 == 0, N % 128 == 0, K % 64 == 0 (holds for all shapes here).
// ---------------------------------------------------------------------------
#define GM 128
#define GN 128
#define GK 64
#define LDT 72   // padded LDS stride in halves (144B, 16B aligned)

__global__ __launch_bounds__(256) void k_gemm(const bf16_t* __restrict__ A,
                                              const bf16_t* __restrict__ W,
                                              int M, int N, int K,
                                              float* __restrict__ outF,
                                              bf16_t* __restrict__ outB,
                                              const float* __restrict__ resid) {
  __shared__ __align__(16) bf16_t As[GM * LDT];
  __shared__ __align__(16) bf16_t Ws[GN * LDT];
  const int tid = threadIdx.x, lane = tid & 31, wv = tid >> 5;
  const int wm = wv >> 2, wn = wv & 3;        // 2 x 4 wave grid
  const int bm = blockIdx.y * GM, bn = blockIdx.x * GN;
  const bf16_t* Ab = A + (size_t)bm * K;
  const bf16_t* Wb = W + (size_t)bn * K;
  v8f acc[4][2];
#pragma unroll
  for (int mt = 0; mt < 4; ++mt)
#pragma unroll
    for (int nt = 0; nt < 2; ++nt) acc[mt][nt] = zero8();

  for (int k0 = 0; k0 < K; k0 += GK) {
    __syncthreads();
    g2l_tile(As, LDT, Ab + k0, (size_t)K, GM, GK >> 3, tid, 256);
    g2l_tile(Ws, LDT, Wb + k0, (size_t)K, GN, GK >> 3, tid, 256);
    if (k0 + GK < K)     // gfx1250 global_prefetch_b8: warm next A tile
      __builtin_prefetch(Ab + (size_t)(tid >> 1) * K + k0 + GK, 0, 1);
    wait_async();
    __syncthreads();
#pragma unroll
    for (int kk = 0; kk < GK; kk += 32) {
      v16bf bfr[2], afr[4];
#pragma unroll
      for (int nt = 0; nt < 2; ++nt)
        bfr[nt] = frag_rowmajor(Ws + (wn * 32 + nt * 16) * LDT + kk, LDT, lane);
#pragma unroll
      for (int mt = 0; mt < 4; ++mt)
        afr[mt] = frag_rowmajor(As + (wm * 64 + mt * 16) * LDT + kk, LDT, lane);
#pragma unroll
      for (int mt = 0; mt < 4; ++mt)
#pragma unroll
        for (int nt = 0; nt < 2; ++nt)
          acc[mt][nt] = wmma_bf16(afr[mt], bfr[nt], acc[mt][nt]);
    }
  }
  // C/D layout: vgpr j -> row (lane<16 ? j : 8+j), col = lane&15
  const int nl = lane & 15;
  const int mh = (lane >> 4) << 3;
#pragma unroll
  for (int mt = 0; mt < 4; ++mt) {
    const int m0 = bm + wm * 64 + mt * 16 + mh;
#pragma unroll
    for (int nt = 0; nt < 2; ++nt) {
      const int n = bn + wn * 32 + nt * 16 + nl;
      if (outF) {
#pragma unroll
        for (int j = 0; j < 8; ++j) {
          size_t i0 = (size_t)(m0 + j) * N + n;
          outF[i0] = acc[mt][nt][j] + (resid ? resid[i0] : 0.f);
        }
      } else {
#pragma unroll
        for (int j = 0; j < 8; ++j)
          outB[(size_t)(m0 + j) * N + n] = f2bf(acc[mt][nt][j]);
      }
    }
  }
}

// ---------------------------------------------------------------------------
// Fused gated FFN GEMM: H = silu(A x W1^T) * (A x W3^T)  -> bf16
// 64x64 tile, 8 waves (2x4), two weight streams, 4 WMMA per 4 fragment loads.
// ---------------------------------------------------------------------------
#define FB 64
#define FK 64

__global__ __launch_bounds__(256) void k_ffn_gate(const bf16_t* __restrict__ A,
                                                  const bf16_t* __restrict__ W1,
                                                  const bf16_t* __restrict__ W3,
                                                  int M, int N, int K,
                                                  bf16_t* __restrict__ H) {
  __shared__ __align__(16) bf16_t As[FB * LDT];
  __shared__ __align__(16) bf16_t W1s[FB * LDT];
  __shared__ __align__(16) bf16_t W3s[FB * LDT];
  const int tid = threadIdx.x, lane = tid & 31, wv = tid >> 5;
  const int wm = wv >> 2, wn = wv & 3;
  const int bm = blockIdx.y * FB, bn = blockIdx.x * FB;
  const bf16_t* Ab  = A  + (size_t)bm * K;
  const bf16_t* W1b = W1 + (size_t)bn * K;
  const bf16_t* W3b = W3 + (size_t)bn * K;
  v8f a10 = zero8(), a11 = zero8(), a30 = zero8(), a31 = zero8();
  for (int k0 = 0; k0 < K; k0 += FK) {
    __syncthreads();
    g2l_tile(As,  LDT, Ab  + k0, (size_t)K, FB, FK >> 3, tid, 256);
    g2l_tile(W1s, LDT, W1b + k0, (size_t)K, FB, FK >> 3, tid, 256);
    g2l_tile(W3s, LDT, W3b + k0, (size_t)K, FB, FK >> 3, tid, 256);
    wait_async();
    __syncthreads();
#pragma unroll
    for (int kk = 0; kk < FK; kk += 32) {
      v16bf b1 = frag_rowmajor(W1s + (wn * 16) * LDT + kk, LDT, lane);
      v16bf b3 = frag_rowmajor(W3s + (wn * 16) * LDT + kk, LDT, lane);
      v16bf a0 = frag_rowmajor(As + (wm * 32) * LDT + kk, LDT, lane);
      v16bf a1 = frag_rowmajor(As + (wm * 32 + 16) * LDT + kk, LDT, lane);
      a10 = wmma_bf16(a0, b1, a10);
      a11 = wmma_bf16(a1, b1, a11);
      a30 = wmma_bf16(a0, b3, a30);
      a31 = wmma_bf16(a1, b3, a31);
    }
  }
  const int n  = bn + wn * 16 + (lane & 15);
  const int mh = (lane >> 4) << 3;
  const int m0 = bm + wm * 32 + mh;
#pragma unroll
  for (int j = 0; j < 8; ++j) {
    float g0 = a10[j], g1 = a11[j];
    float s0 = g0 / (1.f + __expf(-g0));   // silu
    float s1 = g1 / (1.f + __expf(-g1));
    H[(size_t)(m0 + j)      * N + n] = f2bf(s0 * a30[j]);
    H[(size_t)(m0 + 16 + j) * N + n] = f2bf(s1 * a31[j]);
  }
}

// ---------------------------------------------------------------------------
// RoPE in-place on bf16 [B*S, D], head-major interleaved pairs.
// ---------------------------------------------------------------------------
__global__ void k_rope(bf16_t* __restrict__ X, int S, int D, int dk, size_t nPairs) {
  size_t idx = (size_t)blockIdx.x * blockDim.x + threadIdx.x;
  if (idx >= nPairs) return;
  const int halfD = D >> 1;
  size_t row = idx / halfD;
  int p = (int)(idx - row * halfD);       // p = h*(dk/2) + i
  int i = p & ((dk >> 1) - 1);
  int pos = (int)(row & (size_t)(S - 1));
  float ang = (float)pos * __powf(10000.0f, -2.0f * (float)i / (float)dk);
  float sn, cs; __sincosf(ang, &sn, &cs);
  size_t base = row * (size_t)D + ((size_t)p << 1);
  float x1 = bf2f(X[base]), x2 = bf2f(X[base + 1]);
  X[base]     = f2bf(x1 * cs - x2 * sn);
  X[base + 1] = f2bf(x1 * sn + x2 * cs);
}

// ---------------------------------------------------------------------------
// Causal flash attention, bf16 in/out, WMMA QK^T and PV.
// Block = (q-tile 64, head, batch), 4 waves, each wave owns 16 query rows.
// ---------------------------------------------------------------------------
#define AT 64
#define ALD 72

__global__ __launch_bounds__(128) void k_attn(const bf16_t* __restrict__ Q,
                                              const bf16_t* __restrict__ Km,
                                              const bf16_t* __restrict__ V,
                                              bf16_t* __restrict__ ctx,
                                              int S, int D, int dk) {
  __shared__ __align__(16) bf16_t Qs[AT * ALD];
  __shared__ __align__(16) bf16_t Ks[AT * ALD];
  __shared__ __align__(16) bf16_t Vs[AT * ALD];
  __shared__ __align__(16) bf16_t Ps[AT * ALD];   // per-wave 16x64 P slabs
  const int b = blockIdx.z, h = blockIdx.y, q0 = blockIdx.x * AT;
  const int tid = threadIdx.x, lane = tid & 31, wv = tid >> 5;
  const int qr = wv * 16;
  const int mh = (lane >> 4) << 3;
  const int nl = lane & 15;
  const size_t baseOff = (size_t)b * S * D + (size_t)h * dk;
  const bf16_t* Qg = Q  + baseOff;
  const bf16_t* Kg = Km + baseOff;
  const bf16_t* Vg = V  + baseOff;
  g2l_tile(Qs, ALD, Qg + (size_t)q0 * D, (size_t)D, AT, dk >> 3, tid, 128);
  float mrow[8], lrow[8];
#pragma unroll
  for (int j = 0; j < 8; ++j) { mrow[j] = -3.0e38f; lrow[j] = 0.f; }
  v8f o[4];
#pragma unroll
  for (int t = 0; t < 4; ++t) o[t] = zero8();
  const float scale = 0.125f;   // 1/sqrt(dk)

  for (int kt = 0; kt <= q0; kt += AT) {
    __syncthreads();            // guards Ks/Vs reuse (also Qs on 1st iter)
    g2l_tile(Ks, ALD, Kg + (size_t)kt * D, (size_t)D, AT, dk >> 3, tid, 128);
    g2l_tile(Vs, ALD, Vg + (size_t)kt * D, (size_t)D, AT, dk >> 3, tid, 128);
    wait_async();
    __syncthreads();

    // S = (Q K^T) * scale : K tile stored [key, dk] == weight-style [N,K]
    v8f s[4];
#pragma unroll
    for (int t = 0; t < 4; ++t) {
      v8f c = zero8();
#pragma unroll
      for (int kk = 0; kk < 64; kk += 32) {
        v16bf a  = frag_rowmajor(Qs + qr * ALD + kk, ALD, lane);
        v16bf bb = frag_rowmajor(Ks + (t * 16) * ALD + kk, ALD, lane);
        c = wmma_bf16(a, bb, c);
      }
#pragma unroll
      for (int j = 0; j < 8; ++j) c[j] *= scale;
      if (kt == q0) {           // diagonal tile: causal mask
        int key = kt + t * 16 + nl;
#pragma unroll
        for (int j = 0; j < 8; ++j)
          if (key > q0 + qr + mh + j) c[j] = -3.0e38f;
      }
      s[t] = c;
    }

    // online softmax: row stats live per lane-half, shfl within 16 lanes
    float rmax[8], rsum[8];
#pragma unroll
    for (int j = 0; j < 8; ++j) {
      float v = fmaxf(fmaxf(s[0][j], s[1][j]), fmaxf(s[2][j], s[3][j]));
#pragma unroll
      for (int msk = 8; msk >= 1; msk >>= 1) v = fmaxf(v, __shfl_xor(v, msk, 32));
      rmax[j] = v;
    }
#pragma unroll
    for (int j = 0; j < 8; ++j) {
      float mnew  = fmaxf(mrow[j], rmax[j]);
      float alpha = __expf(mrow[j] - mnew);
      mrow[j] = mnew;
      lrow[j] *= alpha;
#pragma unroll
      for (int t = 0; t < 4; ++t) o[t][j] *= alpha;
      rsum[j] = 0.f;
    }
#pragma unroll
    for (int t = 0; t < 4; ++t)
#pragma unroll
      for (int j = 0; j < 8; ++j) {
        float p = __expf(s[t][j] - mrow[j]);
        Ps[(qr + mh + j) * ALD + t * 16 + nl] = f2bf(p);  // wave-local slab
        rsum[j] += p;
      }
#pragma unroll
    for (int j = 0; j < 8; ++j) {
      float v = rsum[j];
#pragma unroll
      for (int msk = 8; msk >= 1; msk >>= 1) v += __shfl_xor(v, msk, 32);
      lrow[j] += v;
    }

    // O += P V : P is A-fragment from this wave's LDS slab (DS in-order),
    // V tile is K-major -> frag_kn.
#pragma unroll
    for (int t = 0; t < 4; ++t) {
#pragma unroll
      for (int kk = 0; kk < AT; kk += 32) {
        v16bf a  = frag_rowmajor(Ps + qr * ALD + kk, ALD, lane);
        v16bf bb = frag_kn(Vs, kk, t * 16, ALD, lane);
        o[t] = wmma_bf16(a, bb, o[t]);
      }
    }
  }

  const size_t orow = (size_t)b * S + q0 + qr + mh;
#pragma unroll
  for (int t = 0; t < 4; ++t)
#pragma unroll
    for (int j = 0; j < 8; ++j)
      ctx[(orow + j) * D + h * dk + t * 16 + nl] = f2bf(o[t][j] / lrow[j]);
}

// ---------------------------------------------------------------------------
// launch
// ---------------------------------------------------------------------------
extern "C" void kernel_launch(void* const* d_in, const int* in_sizes, int n_in,
                              void* d_out, int out_size, void* d_ws, size_t ws_size,
                              hipStream_t stream) {
  (void)in_sizes; (void)n_in; (void)out_size; (void)ws_size;
  const int B = 2, S = 2048, D = 1024, F = 4096, H = 16, dk = 64;
  const int M = B * S;

  const float* x_in = (const float*)d_in[0];
  const float* n1w  = (const float*)d_in[1];
  const float* wq   = (const float*)d_in[2];
  const float* wk   = (const float*)d_in[3];
  const float* wvv  = (const float*)d_in[4];
  const float* wo   = (const float*)d_in[5];
  const float* n2w  = (const float*)d_in[6];
  const float* w1   = (const float*)d_in[7];
  const float* w2   = (const float*)d_in[8];
  const float* w3   = (const float*)d_in[9];
  float* out = (float*)d_out;

  char* ws = (char*)d_ws;
  size_t off = 0;
  auto alloc = [&](size_t bytes) -> void* {
    void* p = ws + off;
    off += (bytes + 255) & ~(size_t)255;
    return p;
  };
  bf16_t* xb    = (bf16_t*)alloc((size_t)M * D * 2);
  bf16_t* wqb   = (bf16_t*)alloc((size_t)D * D * 2);
  bf16_t* wkb   = (bf16_t*)alloc((size_t)D * D * 2);
  bf16_t* wvb   = (bf16_t*)alloc((size_t)D * D * 2);
  bf16_t* wob   = (bf16_t*)alloc((size_t)D * D * 2);
  bf16_t* w1b   = (bf16_t*)alloc((size_t)F * D * 2);
  bf16_t* w3b   = (bf16_t*)alloc((size_t)F * D * 2);
  bf16_t* w2b   = (bf16_t*)alloc((size_t)D * F * 2);
  bf16_t* Qb    = (bf16_t*)alloc((size_t)M * D * 2);
  bf16_t* Kb    = (bf16_t*)alloc((size_t)M * D * 2);
  bf16_t* Vb    = (bf16_t*)alloc((size_t)M * D * 2);
  bf16_t* ctxb  = (bf16_t*)alloc((size_t)M * D * 2);
  float*  after = (float*) alloc((size_t)M * D * 4);
  bf16_t* yb    = (bf16_t*)alloc((size_t)M * D * 2);
  bf16_t* hb    = (bf16_t*)alloc((size_t)M * F * 2);

  // 1. weights -> bf16
  k_cvt<<<dim3(2048), 256, 0, stream>>>(wq,  wqb, (size_t)D * D);
  k_cvt<<<dim3(2048), 256, 0, stream>>>(wk,  wkb, (size_t)D * D);
  k_cvt<<<dim3(2048), 256, 0, stream>>>(wvv, wvb, (size_t)D * D);
  k_cvt<<<dim3(2048), 256, 0, stream>>>(wo,  wob, (size_t)D * D);
  k_cvt<<<dim3(2048), 256, 0, stream>>>(w1,  w1b, (size_t)F * D);
  k_cvt<<<dim3(2048), 256, 0, stream>>>(w3,  w3b, (size_t)F * D);
  k_cvt<<<dim3(2048), 256, 0, stream>>>(w2,  w2b, (size_t)D * F);

  // 2. x = rmsnorm(input) -> bf16
  k_rmsnorm<<<dim3(M), 256, 0, stream>>>(x_in, n1w, xb, D);

  // 3. Q,K,V projections (bf16 out)
  dim3 gP(D / GN, M / GM);     // (8, 32)
  k_gemm<<<gP, 256, 0, stream>>>(xb, wqb, M, D, D, nullptr, Qb, nullptr);
  k_gemm<<<gP, 256, 0, stream>>>(xb, wkb, M, D, D, nullptr, Kb, nullptr);
  k_gemm<<<gP, 256, 0, stream>>>(xb, wvb, M, D, D, nullptr, Vb, nullptr);

  // 4. RoPE on Q and K (in place)
  size_t nPairs = (size_t)M * (D / 2);
  dim3 gR((unsigned)((nPairs + 255) / 256));
  k_rope<<<gR, 256, 0, stream>>>(Qb, S, D, dk, nPairs);
  k_rope<<<gR, 256, 0, stream>>>(Kb, S, D, dk, nPairs);

  // 5. causal flash attention
  dim3 gA(S / AT, H, B);
  k_attn<<<gA, 128, 0, stream>>>(Qb, Kb, Vb, ctxb, S, D, dk);

  // 6. after_mha = ctx @ Wo^T + input  (fp32)
  k_gemm<<<gP, 256, 0, stream>>>(ctxb, wob, M, D, D, after, nullptr, x_in);

  // 7. y = rmsnorm(after_mha) -> bf16
  k_rmsnorm<<<dim3(M), 256, 0, stream>>>(after, n2w, yb, D);

  // 8. h = silu(y @ W1^T) * (y @ W3^T) -> bf16
  dim3 gF(F / FB, M / FB);     // (64, 64)
  k_ffn_gate<<<gF, 256, 0, stream>>>(yb, w1b, w3b, M, F, D, hb);

  // 9. out = h @ W2^T + after_mha  (fp32)
  k_gemm<<<gP, 256, 0, stream>>>(hb, w2b, M, D, F, out, nullptr, after);
}
